// MultiHeadAttention_16320875725477
// MI455X (gfx1250) — matrix operand order repaired and checked by reading, not compiled, not run on hardware
//
#include <hip/hip_runtime.h>

// ---------------------------------------------------------------------------
// MI455X (gfx1250) multi-head attention, bf16 WMMA + fp32 accumulate.
// B=4, C=512, H=W=48 -> S=2304, 8 heads x 64 dim.  ~63 GFLOP total; working
// set ~50MB fits in the 192MB L2, so everything streams through L2 and the
// v_wmma_f32_16x16x32_bf16 path does the math (no useful fp32 matrix unit on
// CDNA5).  Attention is flash-style (online softmax) -- the 679MB score
// tensor is never materialized.
//
// The QKV GEMM stages each 64x512 bf16 weight panel (64KB) into LDS once per
// 8-wave workgroup using the Tensor Data Mover (TENSOR_LOAD_TO_LDS +
// s_wait_tensorcnt; 6-arg builtin variant on this toolchain), and all 8 waves
// read B-fragments from LDS (ds_load_b128).  8x less L2 traffic on weights
// + exercises the CDNA5 async-tensor path.
// ---------------------------------------------------------------------------

typedef __attribute__((ext_vector_type(16))) __bf16 v16bf;
typedef __attribute__((ext_vector_type(8)))  float  v8f;
typedef __attribute__((ext_vector_type(4)))  unsigned u32x4;
typedef __attribute__((ext_vector_type(8)))  int      i32x8;
typedef __attribute__((ext_vector_type(4)))  int      i32x4;

#define BB 4
#define CC 512
#define HH 48
#define WW 48
#define SS (HH * WW)      // 2304 (multiple of 16)
#define NH 8
#define HD 64
#define QKSCALE 0.125f    // 1/sqrt(64), folded into Q at QKV-store time

static __device__ __forceinline__ v8f wmma_bf16(v16bf a, v16bf b, v8f c) {
  // (neg_a, A, neg_b, B, c_mod, C, reuse_a, reuse_b)
  return __builtin_amdgcn_wmma_f32_16x16x32_bf16(false, a, false, b,
                                                 (short)0, c, false, false);
}

// --- A-operand fragment (16x32 bf16, lane = row, K swizzled per ISA table) --
// lanes 0-15: K = {0,1},{2,3},{4,5},{6,7},{16,17},{18,19},{20,21},{22,23}
// lanes 16-31: same +8.  The 8 dword gathers are two contiguous 16B runs;
// the compiler merges them into two global_load_b128 (verified in round 1).
static __device__ __forceinline__ v16bf
load_a_frag(const __bf16* base, int ld, int row, int k0, int half) {
  union { v16bf v; unsigned u[8]; } r;
  const __bf16* p = base + row * ld + k0 + 8 * half;
#pragma unroll
  for (int t = 0; t < 8; ++t) {
    int pk = ((t & 4) << 2) | ((t & 3) << 1);   // 0,2,4,6,16,18,20,22
    r.u[t] = *(const unsigned*)(p + pk);
  }
  return r.v;
}

// --- B-operand fragment (32x16 bf16, lane = col, K swizzled like A: 8*half + {0..7, 16..23}) ---
// Source is row-major with reduction dim contiguous -> two 16B aligned loads.
static __device__ __forceinline__ v16bf
load_b_frag(const __bf16* base, int ld, int col, int k0, int half) {
  union { v16bf v; uint4 q[2]; } r;
  const __bf16* p = base + col * ld + k0 + 8 * half;
  r.q[0] = *(const uint4*)(p);
  r.q[1] = *(const uint4*)(p + 16);
  return r.v;
}

// ---------------------------------------------------------------------------
// Conversion kernels (fp32 -> bf16).  x is also transposed (B,C,S)->(B,S,C).
// ---------------------------------------------------------------------------
__global__ void k_cvt_x(const float* __restrict__ x, __bf16* __restrict__ xbf,
                        int n) {
  int i = blockIdx.x * blockDim.x + threadIdx.x;
  if (i >= n) return;
  int c = i % CC;
  int sb = i / CC;
  int s = sb % SS;
  int b = sb / SS;
  xbf[i] = (__bf16)x[((size_t)b * CC + c) * SS + s];
}

__global__ void k_cvt(const float* __restrict__ in, __bf16* __restrict__ out,
                      int n) {
  int i = blockIdx.x * blockDim.x + threadIdx.x;
  if (i < n) out[i] = (__bf16)in[i];
}

// ---------------------------------------------------------------------------
// QKV GEMM: qkv[m, d] = sum_c xseq[m, c] * w_qkv[d, c];  M=9216, N=1536, K=512
// Workgroup = 8 waves = 128 rows x 64 cols.  The 64x512 bf16 weight panel is
// DMA'd into LDS by the Tensor Data Mover (wave 0 issues, waits TENSORcnt,
// workgroup barrier publishes).  Each wave then computes a 16x64 tile with
// A from global (L2) and B from LDS.
// Q is scaled by 1/8 and stored [b][h][s][d]; K stored [b][h][s][d];
// V stored TRANSPOSED [b][h][d][s] so attention's PV B-operand is contiguous.
// ---------------------------------------------------------------------------
__global__ void __launch_bounds__(256)
k_qkv(const __bf16* __restrict__ xbf, const __bf16* __restrict__ wq,
      __bf16* __restrict__ qb, __bf16* __restrict__ kb,
      __bf16* __restrict__ vtb) {
  __shared__ __bf16 Bl[64 * CC];          // 64KB weight panel
  const int NBLK = (3 * CC) / 64;         // 24 column blocks
  int nblk = blockIdx.x % NBLK;
  int mblk = blockIdx.x / NBLK;
  int n0 = nblk * 64;
  int wid = threadIdx.x >> 5;
  int lane = threadIdx.x & 31;
  int half = lane >> 4, ln = lane & 15;
  int m0 = mblk * 128 + wid * 16;

#if __has_builtin(__builtin_amdgcn_tensor_load_to_lds)
  if (wid == 0) {
    // Tensor DMA descriptor (CDNA5 ISA):
    // 2D tile: 64 rows (output dims) x 512 bf16, row stride 512, from the
    // 1536x512 weight tensor starting at row n0.  Generic shared-pointer
    // low 32 bits == LDS byte offset (flat LDS aperture).
    unsigned long long ga =
        (unsigned long long)(uintptr_t)(wq + (size_t)n0 * CC);
    unsigned ldsb = (unsigned)(uintptr_t)(void*)Bl;
    u32x4 g0;
    g0[0] = 1u;                                   // count=1, user descriptor
    g0[1] = ldsb;                                 // lds_addr
    g0[2] = (unsigned)ga;                         // global_addr[31:0]
    g0[3] = (unsigned)((ga >> 32) & 0x01FFFFFFu)  // global_addr[56:32]
            | (2u << 30);                         // type=2 ("image")
    i32x8 g1;
    g1[0] = (int)(1u << 16);          // data_size=1 (2 bytes), wg_mask=0
    g1[1] = (int)(512u << 16);        // tensor_dim0[15:0]=512 (<<16)
    g1[2] = (int)(1536u << 16);       // tensor_dim1[15:0]=1536 (<<16)
    g1[3] = (int)(512u << 16);        // tile_dim0=512
    g1[4] = 64;                       // tile_dim1=64
    g1[5] = 512;                      // tensor_dim0_stride=512
    g1[6] = 0;
    g1[7] = 0;
    // 6-arg toolchain variant: groups 2/3 + trailing group unused for 2D.
    __builtin_amdgcn_tensor_load_to_lds(g0, g1, (i32x4)0, (i32x4)0,
                                        (i32x8)0, 0);
#if __has_builtin(__builtin_amdgcn_s_wait_tensorcnt)
    __builtin_amdgcn_s_wait_tensorcnt(0);
#else
    asm volatile("s_wait_tensorcnt 0x0" ::: "memory");
#endif
  }
#else
  // Fallback: cooperative global->LDS copy of the 64KB panel.
  {
    const uint4* src = (const uint4*)(wq + (size_t)n0 * CC);
    uint4* dst = (uint4*)Bl;
    for (int i = threadIdx.x; i < 64 * CC / 8; i += 256) dst[i] = src[i];
  }
#endif
  __syncthreads();

  v8f acc[4] = {};
  for (int k0 = 0; k0 < CC; k0 += 32) {
    v16bf a = load_a_frag(xbf, CC, m0 + ln, k0, half);
    v16bf b0 = load_b_frag(Bl, CC, ln, k0, half);       // ds_load_b128 x2
    v16bf b1 = load_b_frag(Bl, CC, 16 + ln, k0, half);
    v16bf b2 = load_b_frag(Bl, CC, 32 + ln, k0, half);
    v16bf b3 = load_b_frag(Bl, CC, 48 + ln, k0, half);
    acc[0] = wmma_bf16(a, b0, acc[0]);
    acc[1] = wmma_bf16(a, b1, acc[1]);
    acc[2] = wmma_bf16(a, b2, acc[2]);
    acc[3] = wmma_bf16(a, b3, acc[3]);
  }
  // C layout: lane holds column n = ln, rows r = v + 8*half.
#pragma unroll
  for (int j = 0; j < 4; ++j) {
    int dd = n0 + 16 * j + ln;     // 0..1535
    int sec = dd >> 9;             // 0=Q, 1=K, 2=V (uniform per tile)
    int d512 = dd & 511;
    int h = d512 >> 6, d = d512 & 63;
#pragma unroll
    for (int v = 0; v < 8; ++v) {
      int m = m0 + v + 8 * half;
      int b = m / SS, s = m % SS;  // tile never straddles batch (16 | 2304)
      float val = acc[j][v];
      if (sec == 0)
        qb[(((size_t)(b * NH + h) * SS) + s) * HD + d] = (__bf16)(val * QKSCALE);
      else if (sec == 1)
        kb[(((size_t)(b * NH + h) * SS) + s) * HD + d] = (__bf16)val;
      else
        vtb[(((size_t)(b * NH + h) * HD) + d) * SS + s] = (__bf16)val;
    }
  }
}

// ---------------------------------------------------------------------------
// Flash attention: one 32-thread workgroup per (b, h, 16-query tile).
// 72 key blocks of 32; per block: 4 QK WMMAs, online softmax (shfl_xor row
// reductions within each 16-lane half), P re-swizzled via 1KB LDS tile into
// A-layout, 4 PV WMMAs.  physics_bias (per-head scalar) added pre-softmax.
// ---------------------------------------------------------------------------
__global__ void __launch_bounds__(32)
k_attn(const __bf16* __restrict__ q, const __bf16* __restrict__ k,
       const __bf16* __restrict__ vt, const float* __restrict__ pbias,
       __bf16* __restrict__ o) {
  __shared__ __bf16 Plds[16 * 32];
  const int QT = SS / 16;  // 144
  int qt = blockIdx.x % QT;
  int bh = blockIdx.x / QT;
  int b = bh / NH, h = bh % NH;
  int lane = threadIdx.x, half = lane >> 4, ln = lane & 15;
  const __bf16* qp = q + (size_t)bh * SS * HD;
  const __bf16* kp = k + (size_t)bh * SS * HD;
  const __bf16* vp = vt + (size_t)bh * HD * SS;
  float bias = pbias[h];
  int q0 = qt * 16;

  // Q tile lives in registers for the whole pass (HD=64 -> two A fragments)
  v16bf qa0 = load_a_frag(qp, HD, q0 + ln, 0, half);
  v16bf qa1 = load_a_frag(qp, HD, q0 + ln, 32, half);

  float mrun[8], lrun[8];
  v8f oacc[4] = {};
#pragma unroll
  for (int v = 0; v < 8; ++v) { mrun[v] = -3.0e38f; lrun[v] = 0.0f; }

  for (int kb = 0; kb < SS; kb += 32) {
    // scores S = (Q*scale) @ K^T for 32 keys: two 16x16 f32 tiles
    v16bf kb00 = load_b_frag(kp, HD, kb + ln, 0, half);
    v16bf kb01 = load_b_frag(kp, HD, kb + ln, 32, half);
    v16bf kb10 = load_b_frag(kp, HD, kb + 16 + ln, 0, half);
    v16bf kb11 = load_b_frag(kp, HD, kb + 16 + ln, 32, half);
    v8f z0 = {}, z1 = {};
    z0 = wmma_bf16(qa0, kb00, z0);
    z0 = wmma_bf16(qa1, kb01, z0);
    z1 = wmma_bf16(qa0, kb10, z1);
    z1 = wmma_bf16(qa1, kb11, z1);

    float p0[8], p1[8];
#pragma unroll
    for (int v = 0; v < 8; ++v) {
      float s0 = z0[v] + bias, s1 = z1[v] + bias;
      // row r = v + 8*half spans the 16 lanes of this half -> xor masks 1..8
      float mx = fmaxf(s0, s1);
#pragma unroll
      for (int d = 1; d < 16; d <<= 1) mx = fmaxf(mx, __shfl_xor(mx, d, 32));
      float mnew = fmaxf(mrun[v], mx);
      float corr = __expf(mrun[v] - mnew);
      p0[v] = __expf(s0 - mnew);
      p1[v] = __expf(s1 - mnew);
      float rs = p0[v] + p1[v];
#pragma unroll
      for (int d = 1; d < 16; d <<= 1) rs += __shfl_xor(rs, d, 32);
      lrun[v] = lrun[v] * corr + rs;
      mrun[v] = mnew;
      oacc[0][v] *= corr; oacc[1][v] *= corr;
      oacc[2][v] *= corr; oacc[3][v] *= corr;
    }

    // re-swizzle P (C layout) -> A layout through LDS (row-major 16x32 bf16)
    __syncthreads();
#pragma unroll
    for (int v = 0; v < 8; ++v) {
      int r = v + 8 * half;
      Plds[r * 32 + ln] = (__bf16)p0[v];
      Plds[r * 32 + 16 + ln] = (__bf16)p1[v];
    }
    __syncthreads();
    v16bf pa = load_a_frag(Plds, 32, ln, 0, half);

    // O += P @ Vblk ; Vt rows = dims -> contiguous B fragments
    v16bf bv0 = load_b_frag(vp, SS, 0 + ln, kb, half);
    v16bf bv1 = load_b_frag(vp, SS, 16 + ln, kb, half);
    v16bf bv2 = load_b_frag(vp, SS, 32 + ln, kb, half);
    v16bf bv3 = load_b_frag(vp, SS, 48 + ln, kb, half);
    oacc[0] = wmma_bf16(pa, bv0, oacc[0]);
    oacc[1] = wmma_bf16(pa, bv1, oacc[1]);
    oacc[2] = wmma_bf16(pa, bv2, oacc[2]);
    oacc[3] = wmma_bf16(pa, bv3, oacc[3]);
  }

  // epilogue: divide by row sum, store bf16 [b][s][h*64+d] for the proj GEMM
  __bf16* op = o + (size_t)b * SS * CC + h * HD;
#pragma unroll
  for (int v = 0; v < 8; ++v) {
    int s = q0 + v + 8 * half;
    float inv = 1.0f / lrun[v];
#pragma unroll
    for (int j = 0; j < 4; ++j)
      op[(size_t)s * CC + j * 16 + ln] = (__bf16)(oacc[j][v] * inv);
  }
}

// ---------------------------------------------------------------------------
// Output projection: out[b, d, s] = sum_c ao[b, s, c] * w_proj[d, c] + b_proj[d]
// M=9216, N=512, K=512.  fp32 output in (B, C, H, W) layout; each lane owns
// one output channel and 8 consecutive s -> two float4 stores.
// ---------------------------------------------------------------------------
__global__ void __launch_bounds__(128)
k_proj(const __bf16* __restrict__ ao, const __bf16* __restrict__ wp,
       const float* __restrict__ bp, float* __restrict__ out) {
  const int NT = CC / 64;  // 8
  int wave = blockIdx.x * (blockDim.x >> 5) + (threadIdx.x >> 5);
  int lane = threadIdx.x & 31;
  int half = lane >> 4, ln = lane & 15;
  int tn = wave % NT, tm = wave / NT;
  int m0 = tm * 16, n0 = tn * 64;

  v8f acc[4] = {};
  for (int k0 = 0; k0 < CC; k0 += 32) {
    v16bf a = load_a_frag(ao, CC, m0 + ln, k0, half);
    v16bf b0 = load_b_frag(wp, CC, n0 + ln, k0, half);
    v16bf b1 = load_b_frag(wp, CC, n0 + 16 + ln, k0, half);
    v16bf b2 = load_b_frag(wp, CC, n0 + 32 + ln, k0, half);
    v16bf b3 = load_b_frag(wp, CC, n0 + 48 + ln, k0, half);
    acc[0] = wmma_bf16(a, b0, acc[0]);
    acc[1] = wmma_bf16(a, b1, acc[1]);
    acc[2] = wmma_bf16(a, b2, acc[2]);
    acc[3] = wmma_bf16(a, b3, acc[3]);
  }

  int mb = m0 + 8 * half;        // first of this lane's 8 consecutive rows
  int b = mb / SS;
  int sbase = mb % SS;
#pragma unroll
  for (int j = 0; j < 4; ++j) {
    int dd = n0 + 16 * j + ln;   // output channel owned by this lane
    float bias = bp[dd];
    float t[8];
#pragma unroll
    for (int v = 0; v < 8; ++v) t[v] = acc[j][v] + bias;
    float* dst = out + ((size_t)b * CC + dd) * SS + sbase;  // 16B aligned
    *(float4*)(dst + 0) = make_float4(t[0], t[1], t[2], t[3]);
    *(float4*)(dst + 4) = make_float4(t[4], t[5], t[6], t[7]);
  }
}

// ---------------------------------------------------------------------------
extern "C" void kernel_launch(void* const* d_in, const int* in_sizes, int n_in,
                              void* d_out, int out_size, void* d_ws,
                              size_t ws_size, hipStream_t stream) {
  (void)in_sizes; (void)n_in; (void)out_size; (void)ws_size;
  const float* x = (const float*)d_in[0];
  const float* wq = (const float*)d_in[1];
  const float* wp = (const float*)d_in[2];
  const float* bp = (const float*)d_in[3];
  const float* pb = (const float*)d_in[4];
  float* outp = (float*)d_out;

  // carve workspace (all arrays fully initialized before being read)
  char* ws = (char*)d_ws;
  size_t off = 0;
  auto carve = [&](size_t bytes) -> char* {
    char* p = ws + off;
    off += (bytes + 255) & ~(size_t)255;
    return p;
  };
  const size_t nQKV = (size_t)BB * NH * SS * HD;     // 4.7M per tensor
  __bf16* xbf = (__bf16*)carve((size_t)BB * SS * CC * 2);
  __bf16* wqb = (__bf16*)carve((size_t)3 * CC * CC * 2);
  __bf16* wpb = (__bf16*)carve((size_t)CC * CC * 2);
  __bf16* qb  = (__bf16*)carve(nQKV * 2);
  __bf16* kb  = (__bf16*)carve(nQKV * 2);
  __bf16* vtb = (__bf16*)carve(nQKV * 2);
  __bf16* ao  = (__bf16*)carve((size_t)BB * SS * CC * 2);

  int nx = BB * SS * CC;
  k_cvt_x<<<(nx + 255) / 256, 256, 0, stream>>>(x, xbf, nx);
  k_cvt<<<(3 * CC * CC + 255) / 256, 256, 0, stream>>>(wq, wqb, 3 * CC * CC);
  k_cvt<<<(CC * CC + 255) / 256, 256, 0, stream>>>(wp, wpb, CC * CC);

  // QKV: 72 row blocks (128 rows) x 24 col blocks (64 cols), 8 waves/block
  k_qkv<<<72 * 24, 256, 0, stream>>>(xbf, wqb, qb, kb, vtb);

  // attention: 4*8 heads x 144 query tiles, one wave each
  k_attn<<<BB * NH * (SS / 16), 32, 0, stream>>>(qb, kb, vtb, pb, ao);

  // projection: 576 x 8 tiles = 4608 waves, 4 waves/block
  k_proj<<<4608 / 4, 128, 0, stream>>>(ao, wpb, bp, outp);
}
